// SparseAttention_55413668053612
// MI455X (gfx1250) — matrix operand order repaired and checked by reading, not compiled, not run on hardware
//
#include <hip/hip_runtime.h>

// Problem constants (fixed by the reference's setup_inputs).
#define BQ   16      // batch
#define NH   32      // query heads
#define NKV  8       // kv heads
#define SEQ  8192    // sequence length
#define DH   128     // head dim
#define HPK  4       // heads per kv head (NH/NKV)
#define KTOP 128     // top-k

typedef __attribute__((ext_vector_type(16))) _Float16 v16h;
typedef __attribute__((ext_vector_type(8)))  float    v8f;

__constant__ const float kScale       = 0.08838834764831843f; // 1/sqrt(128)
__constant__ const float kValidThresh = -32752.0f;            // 0.5 * fp16_min

struct Smem {
  float         sw[HPK][SEQ];           // scores -> weights (128 KB)
  float         wsum[SEQ];              // per-position weight sum over heads (32 KB)
  float         vsel[KTOP][DH];         // gathered selected V rows, async-filled (64 KB)
  unsigned int  validbits[HPK][SEQ/32]; // score > 0.5*fp16_min (4 KB)
  unsigned char mask[SEQ];              // top-k mask (8 KB)
  unsigned char invnib[SEQ];            // bit h = !valid_h (8 KB)
  unsigned int  hist[256];              // radix-select histogram
  unsigned int  chunkcnt[256];          // prefix-scan scratch
  unsigned int  selidx[KTOP];           // selected key indices (ascending)
  _Float16      qh[HPK][DH];            // query in f16
  _Float16      asel[HPK][KTOP];        // adjusted selected weights (f16)
  float         vsum[DH];               // sum over all V rows
  float         vcorr[HPK][DH];         // sum over invalid rows, per head
  float         frall[HPK];             // reallocation value per head
  float         red[8];                 // block reduction scratch (8 waves)
  unsigned int  u0;                     // radix prefix
  unsigned int  u1;                     // remaining-K / needEq
};

__device__ __forceinline__ float wave_sum(float v) {
#pragma unroll
  for (int o = 16; o > 0; o >>= 1) v += __shfl_xor(v, o, 32);
  return v;
}
__device__ __forceinline__ float wave_max(float v) {
#pragma unroll
  for (int o = 16; o > 0; o >>= 1) v = fmaxf(v, __shfl_xor(v, o, 32));
  return v;
}
__device__ __forceinline__ float block_sum(float v, float* red, int tid) {
  v = wave_sum(v);
  __syncthreads();
  if ((tid & 31) == 0) red[tid >> 5] = v;
  __syncthreads();
  float r = 0.0f;
#pragma unroll
  for (int i = 0; i < 8; ++i) r += red[i];
  return r;
}
__device__ __forceinline__ float block_max(float v, float* red, int tid) {
  v = wave_max(v);
  __syncthreads();
  if ((tid & 31) == 0) red[tid >> 5] = v;
  __syncthreads();
  float r = -__builtin_inff();
#pragma unroll
  for (int i = 0; i < 8; ++i) r = fmaxf(r, red[i]);
  return r;
}

__global__ __launch_bounds__(256)
void sparse_attn_gfx1250(const float* __restrict__ q,
                         const float* __restrict__ kc,
                         const float* __restrict__ vc,
                         const float* __restrict__ lm,
                         float* __restrict__ out,     // (B, NH, 1, DH)
                         float* __restrict__ wout) {  // (B, NH, 1, SEQ)
  extern __shared__ char smem_raw[];
  Smem& sm = *reinterpret_cast<Smem*>(smem_raw);

  const int tid  = threadIdx.x;
  const int lane = tid & 31;
  const int wid  = tid >> 5;     // 8 waves
  const int bg   = blockIdx.x;   // one workgroup per (b, kv-head)
  const int b    = bg / NKV;
  const int g    = bg % NKV;

  const float* qbase  = q  + (size_t)(b * NH + g * HPK) * DH;
  const float* kbase  = kc + ((size_t)(b * NKV + g)) * SEQ * DH;
  const float* vbase  = vc + ((size_t)(b * NKV + g)) * SEQ * DH;
  const float* lmbase = lm + (size_t)(b * NH + g * HPK) * SEQ;
  float* obase = out  + (size_t)(b * NH + g * HPK) * DH;
  float* wbase = wout + (size_t)(b * NH + g * HPK) * SEQ;

  // ---- stage q (f16) to LDS, clear validity bits and V accumulators ----
  for (int i = tid; i < HPK * DH; i += 256)
    ((_Float16*)sm.qh)[i] = (_Float16)qbase[i];
  for (int i = tid; i < HPK * (SEQ / 32); i += 256)
    ((unsigned int*)sm.validbits)[i] = 0u;
  for (int i = tid; i < DH; i += 256) sm.vsum[i] = 0.0f;
  for (int i = tid; i < HPK * DH; i += 256) ((float*)sm.vcorr)[i] = 0.0f;
  __syncthreads();

  // A operand k order chosen here (contiguous, not AMD's table order): lanes 0-15 -> K 0..15; lanes 16-31 -> K 16..31 (sequential, matches B)
  const int arow = (lane & 15) & 3;            // replicate heads into rows 4..15 (ignored)
  const int koff = (lane >= 16) ? 16 : 0;
  // B operand k order chosen here (same as A): lanes 0-15 -> K 0..15; lanes 16-31 -> K 16..31 (sequential)
  const int khalf = (lane >= 16) ? 16 : 0;

  // ---------------- Phase 1: scores = scale * q @ K^T + logmask ----------------
  v16h afrag[4];
#pragma unroll
  for (int c = 0; c < 4; ++c) {
    const int base = c * 32 + koff;
#pragma unroll
    for (int i = 0; i < 8; ++i) {
      afrag[c][i]     = sm.qh[arow][base + i];
      afrag[c][i + 8] = sm.qh[arow][base + 8 + i];
    }
  }

  for (int t = wid; t < SEQ / 16; t += 8) {      // 64 tiles of 16 keys per wave
    const int key = t * 16 + (lane & 15);
    const float* krow = kbase + (size_t)key * DH;
    if (t + 8 < SEQ / 16)
      __builtin_prefetch(krow + 128 * DH, 0, 1); // global_prefetch for next tile
    v8f acc = {};
#pragma unroll
    for (int c = 0; c < 4; ++c) {
      const int base = c * 32 + khalf;
      const float4 f0 = *(const float4*)(krow + base);
      const float4 f1 = *(const float4*)(krow + base + 4);
      const float4 f2 = *(const float4*)(krow + base + 8);
      const float4 f3 = *(const float4*)(krow + base + 12);
      v16h bfrag;
      bfrag[0]  = (_Float16)f0.x; bfrag[1]  = (_Float16)f0.y;
      bfrag[2]  = (_Float16)f0.z; bfrag[3]  = (_Float16)f0.w;
      bfrag[4]  = (_Float16)f1.x; bfrag[5]  = (_Float16)f1.y;
      bfrag[6]  = (_Float16)f1.z; bfrag[7]  = (_Float16)f1.w;
      bfrag[8]  = (_Float16)f2.x; bfrag[9]  = (_Float16)f2.y;
      bfrag[10] = (_Float16)f2.z; bfrag[11] = (_Float16)f2.w;
      bfrag[12] = (_Float16)f3.x; bfrag[13] = (_Float16)f3.y;
      bfrag[14] = (_Float16)f3.z; bfrag[15] = (_Float16)f3.w;
      acc = __builtin_amdgcn_wmma_f32_16x16x32_f16(false, afrag[c], false, bfrag,
                                                   (short)0, acc, false, false);
    }
    if (lane < 16) {                 // C layout: lanes 0-15, VGPR r = row r, col = lane
#pragma unroll
      for (int r = 0; r < HPK; ++r) {
        const float s = acc[r] * kScale + lmbase[(size_t)r * SEQ + key];
        sm.sw[r][key] = s;
        if (s > kValidThresh)
          atomicOr(&sm.validbits[r][key >> 5], 1u << (key & 31));
      }
    }
  }
  __syncthreads();

  // invalid-nibble per position (bit h set => head h invalid at j)
  for (int j = tid; j < SEQ; j += 256) {
    unsigned int nib = 0;
#pragma unroll
    for (int h = 0; h < HPK; ++h)
      nib |= ((((sm.validbits[h][j >> 5] >> (j & 31)) & 1u) ^ 1u) << h);
    sm.invnib[j] = (unsigned char)nib;
  }

  // ---------------- Phase 2a: fp32 softmax per head ----------------
  for (int h = 0; h < HPK; ++h) {
    float m = -__builtin_inff();
    for (int j = tid; j < SEQ; j += 256) m = fmaxf(m, sm.sw[h][j]);
    m = block_max(m, sm.red, tid);
    float ssum = 0.0f;
    for (int j = tid; j < SEQ; j += 256) {
      const float e = __expf(sm.sw[h][j] - m);
      sm.sw[h][j] = e;
      ssum += e;
    }
    ssum = block_sum(ssum, sm.red, tid);
    const float inv = 1.0f / ssum;
    for (int j = tid; j < SEQ; j += 256) sm.sw[h][j] *= inv;
    __syncthreads();
  }

  // wsum[j] = sum over the 4 heads (basis for top-k)
  for (int j = tid; j < SEQ; j += 256)
    sm.wsum[j] = sm.sw[0][j] + sm.sw[1][j] + sm.sw[2][j] + sm.sw[3][j];
  if (tid == 0) { sm.u0 = 0u; sm.u1 = KTOP; }
  __syncthreads();

  // ---------------- Phase 2b: top-128 radix select on wsum ----------------
  // wsum >= 0, so uint bit pattern preserves float ordering.
  for (int level = 3; level >= 0; --level) {
    const int shift = level * 8;
    sm.hist[tid] = 0;
    __syncthreads();
    const unsigned int pmask = (level == 3) ? 0u : (0xFFFFFFFFu << ((level + 1) * 8));
    const unsigned int pref  = sm.u0;
    for (int j = tid; j < SEQ; j += 256) {
      const unsigned int bits = __float_as_uint(sm.wsum[j]);
      if ((bits & pmask) == pref)
        atomicAdd(&sm.hist[(bits >> shift) & 255], 1u);
    }
    __syncthreads();
    if (tid == 0) {
      const unsigned int rem = sm.u1;
      unsigned int cum = 0;
      int bsel = 0;
      for (int bin = 255; bin >= 0; --bin) {
        const unsigned int c = sm.hist[bin];
        if (cum + c >= rem) { bsel = bin; break; }
        cum += c;
      }
      sm.u0 |= ((unsigned int)bsel) << shift;
      sm.u1 = rem - cum;
    }
    __syncthreads();
  }
  const unsigned int T      = sm.u0;  // value of the 128th-largest wsum
  const unsigned int needEq = sm.u1;  // ties at T to take (lowest indices, like lax.top_k)

  // Build mask: strictly-greater, plus lowest-index ties.
  const int CH = SEQ / 256;           // 32 contiguous elements per thread
  const int j0 = tid * CH;
  unsigned int eqcnt = 0;
  for (int j = j0; j < j0 + CH; ++j) {
    const unsigned int bits = __float_as_uint(sm.wsum[j]);
    sm.mask[j] = (bits > T) ? 1 : 0;
    if (bits == T) eqcnt++;
  }
  sm.chunkcnt[tid] = eqcnt;
  __syncthreads();
  if (tid == 0) {
    unsigned int run = 0;
    for (int i = 0; i < 256; ++i) { const unsigned int c = sm.chunkcnt[i]; sm.chunkcnt[i] = run; run += c; }
  }
  __syncthreads();
  unsigned int rank = sm.chunkcnt[tid];
  for (int j = j0; j < j0 + CH; ++j) {
    if (__float_as_uint(sm.wsum[j]) == T) {
      if (rank < needEq) sm.mask[j] = 1;
      rank++;
    }
  }
  __syncthreads();

  // ---- compact selected indices (exactly KTOP, ascending) ----
  unsigned int mcnt = 0;
  for (int j = j0; j < j0 + CH; ++j) mcnt += sm.mask[j];
  sm.chunkcnt[tid] = mcnt;
  __syncthreads();
  if (tid == 0) {
    unsigned int run = 0;
    for (int i = 0; i < 256; ++i) { const unsigned int c = sm.chunkcnt[i]; sm.chunkcnt[i] = run; run += c; }
  }
  __syncthreads();
  unsigned int pos = sm.chunkcnt[tid];
  for (int j = j0; j < j0 + CH; ++j)
    if (sm.mask[j]) sm.selidx[pos++] = (unsigned int)j;
  __syncthreads();

  // ---- async gather of the 128 selected V rows into LDS (overlaps phases below) ----
  // 256 threads x 16 x b128 = 64 KB, tracked by ASYNCcnt.
  for (int r = 0; r < 16; ++r) {
    const int id  = tid + 256 * r;          // 0..4095 16B transfers
    const int row = id >> 5;                // 32 transfers per 512B row
    const int col = (id & 31) * 4;
    const unsigned long long gaddr =
        (unsigned long long)(uintptr_t)(vbase + (size_t)sm.selidx[row] * DH + col);
    const unsigned int ldsoff = (unsigned int)(uintptr_t)(const void*)&sm.vsel[row][col];
    asm volatile("global_load_async_to_lds_b128 %0, %1, off"
                 :: "v"(ldsoff), "v"(gaddr) : "memory");
  }

  // ---------------- Phase 2c: mask + reallocate leftover mass; emit weights ----------------
  for (int h = 0; h < HPK; ++h) {
    float msum = 0.0f, rcntf = 0.0f;
    for (int j = tid; j < SEQ; j += 256) {
      const bool mk = sm.mask[j] != 0;
      const bool vl = (sm.validbits[h][j >> 5] >> (j & 31)) & 1u;
      if (mk)      msum  += sm.sw[h][j];
      else if (vl) rcntf += 1.0f;
    }
    msum  = block_sum(msum,  sm.red, tid);
    rcntf = block_sum(rcntf, sm.red, tid);
    const float ralloc = (rcntf > 0.0f) ? (1.0f - msum) / rcntf : 0.0f;
    if (tid == 0) sm.frall[h] = ralloc;
    for (int j = tid; j < SEQ; j += 256) {
      const bool mk = sm.mask[j] != 0;
      const bool vl = (sm.validbits[h][j >> 5] >> (j & 31)) & 1u;
      const float wv = mk ? sm.sw[h][j] : (vl ? ralloc : 0.0f);
      sm.sw[h][j] = wv;
      wbase[(size_t)h * SEQ + j] = wv;   // weights output, coalesced
    }
    __syncthreads();
  }

  // ---------------- Phase 3a: dense V stream: Vsum and per-head invalid correction ----------------
  {
    const int d0 = lane * 4;               // each lane owns 4 contiguous dims
    float4 vs = make_float4(0.f, 0.f, 0.f, 0.f);
    const int jbeg = wid * (SEQ / 8), jend = jbeg + (SEQ / 8);
    for (int j = jbeg; j < jend; ++j) {
      const float4 vv = *(const float4*)(vbase + (size_t)j * DH + d0);
      if (j + 64 < SEQ)
        __builtin_prefetch(vbase + (size_t)(j + 64) * DH + d0, 0, 1);
      const unsigned int nib = sm.invnib[j];
      vs.x += vv.x; vs.y += vv.y; vs.z += vv.z; vs.w += vv.w;
      if (nib) {                            // rare: only where logmask invalidated scores
#pragma unroll
        for (int h = 0; h < HPK; ++h)
          if ((nib >> h) & 1u) {
            atomicAdd(&sm.vcorr[h][d0 + 0], vv.x);
            atomicAdd(&sm.vcorr[h][d0 + 1], vv.y);
            atomicAdd(&sm.vcorr[h][d0 + 2], vv.z);
            atomicAdd(&sm.vcorr[h][d0 + 3], vv.w);
          }
      }
    }
    atomicAdd(&sm.vsum[d0 + 0], vs.x);
    atomicAdd(&sm.vsum[d0 + 1], vs.y);
    atomicAdd(&sm.vsum[d0 + 2], vs.z);
    atomicAdd(&sm.vsum[d0 + 3], vs.w);
  }
  __syncthreads();

  // ---- stage adjusted selected weights: A[h][i] = w[h][sel_i] - r_h * valid_h[sel_i] ----
  for (int i = tid; i < HPK * KTOP; i += 256) {
    const int h = i >> 7;
    const int s = i & (KTOP - 1);
    const unsigned int j = sm.selidx[s];
    const float vl = ((sm.validbits[h][j >> 5] >> (j & 31)) & 1u) ? 1.0f : 0.0f;
    sm.asel[h][s] = (_Float16)(sm.sw[h][j] - sm.frall[h] * vl);
  }
  asm volatile("s_wait_asynccnt 0" ::: "memory");  // our gathered V rows are in LDS
  __syncthreads();

  // ---------------- Phase 3b: out = A_sel @ V_sel (WMMA) + r_h * (Vsum - Vcorr_h) ----------------
  {
    const int dcol = wid * 16 + (lane & 15);  // 8 waves cover D = 8 tiles of 16 dims
    v8f acc = {};
#pragma unroll
    for (int kcn = 0; kcn < KTOP / 32; ++kcn) {   // 4 chunks of 32 selected keys
      v16h af, bf;
      const int ab = kcn * 32 + koff;
#pragma unroll
      for (int i = 0; i < 8; ++i) {
        af[i]     = sm.asel[arow][ab + i];
        af[i + 8] = sm.asel[arow][ab + 8 + i];
      }
      const int bb = kcn * 32 + khalf;
#pragma unroll
      for (int i = 0; i < 16; ++i)
        bf[i] = (_Float16)sm.vsel[bb + i][dcol];
      acc = __builtin_amdgcn_wmma_f32_16x16x32_f16(false, af, false, bf,
                                                   (short)0, acc, false, false);
    }
    if (lane < 16) {
#pragma unroll
      for (int r = 0; r < HPK; ++r) {
        const float extra = sm.frall[r] * (sm.vsum[dcol] - sm.vcorr[r][dcol]);
        obase[(size_t)r * DH + dcol] = acc[r] + extra;
      }
    }
  }
}

extern "C" void kernel_launch(void* const* d_in, const int* in_sizes, int n_in,
                              void* d_out, int out_size, void* d_ws, size_t ws_size,
                              hipStream_t stream) {
  (void)in_sizes; (void)n_in; (void)out_size; (void)d_ws; (void)ws_size;
  const float* q  = (const float*)d_in[0];
  const float* k  = (const float*)d_in[1];
  const float* v  = (const float*)d_in[2];
  const float* lm = (const float*)d_in[3];
  float* out  = (float*)d_out;                       // (B,NH,1,DH) first
  float* wout = out + (size_t)BQ * NH * DH;          // then (B,NH,1,SEQ)

  dim3 grid(BQ * NKV);   // 128 workgroups, one per (batch, kv-head)
  dim3 block(256);       // 8 wave32 waves
  hipLaunchKernelGGL(sparse_attn_gfx1250, grid, block, sizeof(Smem), stream,
                     q, k, v, lm, out, wout);
}
